// GATEmbedding_48747878809938
// MI455X (gfx1250) — matrix operand order repaired and checked by reading, not compiled, not run on hardware
//
#include <hip/hip_runtime.h>
#include <hip/hip_bf16.h>
#include <math.h>

// Problem constants (match reference)
#define B_   2
#define N_   512
#define D_   128
#define DE_  32
#define NEG_SLOPE 0.2f
#define NEG_BIG   (-1e30f)
#define JCHUNK 128              // j range handled per block of kernel 2
#define NJC    4                // N_/JCHUNK
#define NIT    32               // N_/16 i-tiles

typedef __bf16 bf16_t;
typedef __attribute__((ext_vector_type(16))) __bf16 v16bf;
typedef __attribute__((ext_vector_type(8)))  float  v8f;

__device__ __forceinline__ v8f wmma_bf16(v16bf a, v16bf b, v8f c) {
  // D = A(16x32 bf16) * B(32x16 bf16) + C(16x16 f32)
  return __builtin_amdgcn_wmma_f32_16x16x32_bf16(false, a, false, b, (short)0, c,
                                                 false, false);
}

// Pack four float4's (16 consecutive-by-layout values) into a bf16 A/B fragment.
__device__ __forceinline__ v16bf pack_bf16(float4 a, float4 b, float4 c, float4 d) {
  v16bf v;
  v[0]=(bf16_t)a.x;  v[1]=(bf16_t)a.y;  v[2]=(bf16_t)a.z;  v[3]=(bf16_t)a.w;
  v[4]=(bf16_t)b.x;  v[5]=(bf16_t)b.y;  v[6]=(bf16_t)b.z;  v[7]=(bf16_t)b.w;
  v[8]=(bf16_t)c.x;  v[9]=(bf16_t)c.y;  v[10]=(bf16_t)c.z; v[11]=(bf16_t)c.w;
  v[12]=(bf16_t)d.x; v[13]=(bf16_t)d.y; v[14]=(bf16_t)d.z; v[15]=(bf16_t)d.w;
  return v;
}

// ---------------------------------------------------------------------------
// Kernel 1: xl = x@W_l + b_l ; xr = x@W_r + b_r     (wave0 -> xl, wave1 -> xr)
// One block per (batch, 16-node tile), 2 waves.
// ---------------------------------------------------------------------------
__global__ void __launch_bounds__(64)
node_transform_kernel(const float* __restrict__ x,
                      const float* __restrict__ Wl, const float* __restrict__ bl,
                      const float* __restrict__ Wr, const float* __restrict__ br,
                      float* __restrict__ xl, float* __restrict__ xr)
{
  const int blk  = blockIdx.x;          // b*NIT + it
  const int b    = blk / NIT;
  const int i0   = (blk % NIT) * 16;
  const int wave = threadIdx.x >> 5;
  const int l    = threadIdx.x & 31;
  const int h    = l >> 4;              // half of the wave
  const int nl   = l & 15;

  const float* W   = wave ? Wr : Wl;
  const float* bv  = wave ? br : bl;
  float*       dst = wave ? xr : xl;

  // A fragments: x tile [16 x 128] in 4 chunks of K=32.
  // Lane layout: row m = nl; ks = {h*8..h*8+7, 16+h*8..16+h*8+7} per chunk.
  v16bf a[4];
  const float* xrow = x + ((size_t)(b * N_ + i0 + nl)) * D_;
#pragma unroll
  for (int kc = 0; kc < 4; ++kc) {
    const float* p = xrow + kc * 32 + h * 8;
    a[kc] = pack_bf16(*reinterpret_cast<const float4*>(p),
                      *reinterpret_cast<const float4*>(p + 4),
                      *reinterpret_cast<const float4*>(p + 16),
                      *reinterpret_cast<const float4*>(p + 20));
  }

#pragma unroll
  for (int nt = 0; nt < 8; ++nt) {      // 8 output n-tiles of 16 cols
    v8f acc = {};
#pragma unroll
    for (int kc = 0; kc < 4; ++kc) {
      // B fragment: col n = nt*16+nl, ks = {h*8..h*8+7, 16+h*8..16+h*8+7} within chunk
      v16bf bw;
#pragma unroll
      for (int c = 0; c < 16; ++c)
        bw[c] = (bf16_t)W[(size_t)(kc * 32 + h * 8 + c + (c >= 8 ? 8 : 0)) * D_ + nt * 16 + nl];
      acc = wmma_bf16(a[kc], bw, acc);
    }
    const float bb = bv[nt * 16 + nl];
#pragma unroll
    for (int r = 0; r < 8; ++r) {
      const int m = h * 8 + r;          // C layout: row = h*8+r, col = nl
      dst[((size_t)(b * N_ + i0 + m)) * D_ + nt * 16 + nl] = acc[r] + bb;
    }
  }
}

// ---------------------------------------------------------------------------
// Kernel 2: fused edge transform + LeakyReLU + att-dot + masked online softmax
//           + weighted aggregation of xl. One block per (b, i-tile, j-chunk).
// 8 waves; wave w handles j = j0 + w, w+8, ...  Partials (m, l, acc) -> ws.
// The x_r tile rides in the WMMA C operand (it is exactly the C layout), so
// the matrix engine performs the `+ x_r[i]` of GATv2's `pre` for free.
// ---------------------------------------------------------------------------
__global__ void __launch_bounds__(256)
edge_attn_partial_kernel(const float* __restrict__ e,
                         const unsigned char* __restrict__ adj,
                         const float* __restrict__ We,
                         const float* __restrict__ att,
                         const float* __restrict__ xl,
                         const float* __restrict__ xr,
                         float* __restrict__ pm, float* __restrict__ pl,
                         float* __restrict__ pacc)
{
  __shared__ float s_acc[16 * D_];            // 8 KB shared accumulator
  __shared__ float s_m[16], s_l[16];
  __shared__ float wm[8][16], wls[8][16];
  __shared__ unsigned char s_adj[16 * JCHUNK];

  const int blk = blockIdx.x;                 // (b*NIT + it)*NJC + jc
  const int jc  = blk % NJC;
  const int it  = (blk / NJC) % NIT;
  const int b   = blk / (NJC * NIT);
  const int i0  = it * 16, j0 = jc * JCHUNK;

  const int tid = threadIdx.x;
  const int w = tid >> 5, l = tid & 31, h = l >> 4, nl = l & 15;

  for (int idx = tid; idx < 16 * D_; idx += 256) s_acc[idx] = 0.f;
  for (int idx = tid; idx < 16 * JCHUNK; idx += 256) {
    const int m = idx >> 7, jj = idx & (JCHUNK - 1);
    s_adj[idx] = adj[((size_t)(b * N_ + i0 + m)) * N_ + j0 + jj];
  }
  __syncthreads();

  // Resident: W_e B-fragments (32x128 -> 8 n-tiles), att values,
  // and the x_r tile pre-staged as WMMA C-fragments.
  v16bf we[8];
  float attv[8];
  v8f cxr[8];
#pragma unroll
  for (int nt = 0; nt < 8; ++nt) {
#pragma unroll
    for (int c = 0; c < 16; ++c)
      we[nt][c] = (bf16_t)We[(size_t)(h * 8 + c + (c >= 8 ? 8 : 0)) * D_ + nt * 16 + nl];
    attv[nt] = att[nt * 16 + nl];
#pragma unroll
    for (int r = 0; r < 8; ++r)
      cxr[nt][r] = xr[((size_t)(b * N_ + i0 + h * 8 + r)) * D_ + nt * 16 + nl];
  }

  // Online-softmax state: rows h*8+r live in this lane's half-group.
  float fm[8], fl[8];
  v8f accf[8];
#pragma unroll
  for (int r = 0; r < 8; ++r) { fm[r] = NEG_BIG; fl[r] = 0.f; }
#pragma unroll
  for (int nt = 0; nt < 8; ++nt) { v8f z = {}; accf[nt] = z; }

  for (int jj = w; jj < JCHUNK; jj += 8) {
    const int j = j0 + jj;
    // A fragment from e[b, i0+m, j, 0..31] (row m = nl, ks per bf16 A layout)
    const float* ep = e + (((size_t)(b * N_ + i0 + nl)) * N_ + j) * DE_;
    if (jj + 8 < JCHUNK) __builtin_prefetch(ep + 8 * DE_, 0, 0);
    const float* p = ep + h * 8;
    v16bf ae = pack_bf16(*reinterpret_cast<const float4*>(p),
                         *reinterpret_cast<const float4*>(p + 4),
                         *reinterpret_cast<const float4*>(p + 16),
                         *reinterpret_cast<const float4*>(p + 20));

    float xlv[8];
#pragma unroll
    for (int nt = 0; nt < 8; ++nt)
      xlv[nt] = xl[((size_t)(b * N_ + j)) * D_ + nt * 16 + nl];

    // pre = e@We + x_r (in C) + x_l; then LeakyReLU(pre) . att reduced over d.
    float s[8];
#pragma unroll
    for (int r = 0; r < 8; ++r) s[r] = 0.f;
#pragma unroll
    for (int nt = 0; nt < 8; ++nt) {
      v8f et = wmma_bf16(ae, we[nt], cxr[nt]);   // C = x_r fragment
#pragma unroll
      for (int r = 0; r < 8; ++r) {
        float pr = et[r] + xlv[nt];
        pr = fmaxf(pr, NEG_SLOPE * pr);          // LeakyReLU (slope > 0)
        s[r] = fmaf(pr, attv[nt], s[r]);
      }
    }
    // Reduce over the 16 lanes of each half (cols of this lane's rows).
#pragma unroll
    for (int off = 1; off < 16; off <<= 1) {
#pragma unroll
      for (int r = 0; r < 8; ++r) s[r] += __shfl_xor(s[r], off, 16);
    }

    // Masked online softmax update.
    float ps[8], sc[8];
#pragma unroll
    for (int r = 0; r < 8; ++r) {
      const int  m  = h * 8 + r;
      const bool ab = s_adj[m * JCHUNK + jj] || ((i0 + m) == j);  // +self-loop
      const float sv   = ab ? s[r] : NEG_BIG;
      const float mnew = fmaxf(fm[r], sv);
      sc[r] = __expf(fm[r] - mnew);
      ps[r] = ab ? __expf(sv - mnew) : 0.f;
      fl[r] = fl[r] * sc[r] + ps[r];
      fm[r] = mnew;
    }
    // acc = acc*scale + p * xl[j,:]   (rank-1 update in C-fragment layout)
#pragma unroll
    for (int nt = 0; nt < 8; ++nt) {
#pragma unroll
      for (int r = 0; r < 8; ++r)
        accf[nt][r] = fmaf(accf[nt][r], sc[r], ps[r] * xlv[nt]);
    }
  }

  // Cross-wave combine in LDS.
  if (nl == 0) {
#pragma unroll
    for (int r = 0; r < 8; ++r) { wm[w][h * 8 + r] = fm[r]; wls[w][h * 8 + r] = fl[r]; }
  }
  __syncthreads();
  if (tid < 16) {
    float M = NEG_BIG;
    for (int ww = 0; ww < 8; ++ww) M = fmaxf(M, wm[ww][tid]);
    float L = 0.f;
    for (int ww = 0; ww < 8; ++ww) L += wls[ww][tid] * __expf(wm[ww][tid] - M);
    s_m[tid] = M; s_l[tid] = L;
  }
  __syncthreads();
#pragma unroll
  for (int r = 0; r < 8; ++r) {
    const int m = h * 8 + r;
    const float f = __expf(fm[r] - s_m[m]);
#pragma unroll
    for (int nt = 0; nt < 8; ++nt)
      atomicAdd(&s_acc[m * D_ + nt * 16 + nl], accf[nt][r] * f);   // ds_add_f32
  }
  __syncthreads();

  // Emit per-block partial (m, l, unnormalized acc) to workspace.
  if (tid < 16) {
    pm[(size_t)blk * 16 + tid] = s_m[tid];
    pl[(size_t)blk * 16 + tid] = s_l[tid];
  }
  float* pap = pacc + (size_t)blk * (16 * D_);
  for (int idx = tid; idx < 16 * D_; idx += 256) pap[idx] = s_acc[idx];
}

// ---------------------------------------------------------------------------
// Kernel 3: merge the NJC j-chunk partials, normalize, +bias, mask, SiLU.
// ---------------------------------------------------------------------------
__global__ void __launch_bounds__(256)
combine_silu_kernel(const float* __restrict__ pm, const float* __restrict__ pl,
                    const float* __restrict__ pacc,
                    const float* __restrict__ bias,
                    const unsigned char* __restrict__ mask,
                    float* __restrict__ out)
{
  const int blk = blockIdx.x;             // b*NIT + it
  const int b = blk / NIT, it = blk % NIT, i0 = it * 16;
  const int base = blk * NJC;

  __shared__ float Mrow[16], Lrow[16];
  if (threadIdx.x < 16) {
    float M = NEG_BIG;
    for (int c = 0; c < NJC; ++c)
      M = fmaxf(M, pm[(size_t)(base + c) * 16 + threadIdx.x]);
    float L = 0.f;
    for (int c = 0; c < NJC; ++c)
      L += pl[(size_t)(base + c) * 16 + threadIdx.x] *
           __expf(pm[(size_t)(base + c) * 16 + threadIdx.x] - M);
    Mrow[threadIdx.x] = M; Lrow[threadIdx.x] = L;
  }
  __syncthreads();

  for (int idx = threadIdx.x; idx < 16 * D_; idx += 256) {
    const int row = idx >> 7, d = idx & (D_ - 1);
    float A = 0.f;
    for (int c = 0; c < NJC; ++c) {
      const float f = __expf(pm[(size_t)(base + c) * 16 + row] - Mrow[row]);
      A = fmaf(pacc[(size_t)(base + c) * (16 * D_) + idx], f, A);
    }
    float o = A / Lrow[row] + bias[d];
    o = o / (1.f + __expf(-o));                         // SiLU
    o = mask[b * N_ + i0 + row] ? o : 0.f;
    out[((size_t)(b * N_ + i0 + row)) * D_ + d] = o;
  }
}

// ---------------------------------------------------------------------------
extern "C" void kernel_launch(void* const* d_in, const int* in_sizes, int n_in,
                              void* d_out, int out_size, void* d_ws, size_t ws_size,
                              hipStream_t stream) {
  const float*         x    = (const float*)d_in[0];
  const unsigned char* adj  = (const unsigned char*)d_in[1];   // bool (1B)
  const float*         e    = (const float*)d_in[2];
  const unsigned char* mask = (const unsigned char*)d_in[3];   // bool (1B)
  const float*         Wl   = (const float*)d_in[4];
  const float*         bl   = (const float*)d_in[5];
  const float*         Wr   = (const float*)d_in[6];
  const float*         br   = (const float*)d_in[7];
  const float*         We   = (const float*)d_in[8];
  const float*         att  = (const float*)d_in[9];
  const float*         bias = (const float*)d_in[10];
  float* out = (float*)d_out;

  // Workspace carve-up (~3.1 MB total).
  char* ws = (char*)d_ws;
  float* xl = (float*)ws;  ws += (size_t)B_ * N_ * D_ * sizeof(float);
  float* xr = (float*)ws;  ws += (size_t)B_ * N_ * D_ * sizeof(float);
  float* pm = (float*)ws;  ws += (size_t)B_ * NIT * NJC * 16 * sizeof(float);
  float* pl = (float*)ws;  ws += (size_t)B_ * NIT * NJC * 16 * sizeof(float);
  float* pacc = (float*)ws; // B_*NIT*NJC * 16*D_ floats = 2 MB

  node_transform_kernel<<<B_ * NIT, 64, 0, stream>>>(x, Wl, bl, Wr, br, xl, xr);
  edge_attn_partial_kernel<<<B_ * NIT * NJC, 256, 0, stream>>>(
      e, adj, We, att, xl, xr, pm, pl, pacc);
  combine_silu_kernel<<<B_ * NIT, 256, 0, stream>>>(pm, pl, pacc, bias, mask, out);
}
